// GCN_1_hidden_model_70145405878898
// MI455X (gfx1250) — compile-verified, matrix operand order checked by reading, not run on hardware
//
#include <hip/hip_runtime.h>
#include <stdint.h>

// ---------------------------------------------------------------------------
// Types for CDNA5 WMMA (wave32): V_WMMA_F32_16X16X32_BF16
// ---------------------------------------------------------------------------
typedef __bf16 bf16_t;
typedef __bf16 v8bf  __attribute__((ext_vector_type(8)));
typedef __bf16 v16bf __attribute__((ext_vector_type(16)));
typedef float  v8f   __attribute__((ext_vector_type(8)));

__device__ __forceinline__ bf16_t f2bf(float f) {
  // round-to-nearest-even fp32 -> bf16
  union { float f; unsigned u; } in; in.f = f;
  unsigned u = in.u;
  u += 0x7FFFu + ((u >> 16) & 1u);
  union { unsigned short s; bf16_t b; } out; out.s = (unsigned short)(u >> 16);
  return out.b;
}

// ---------------------------------------------------------------------------
// Small elementwise helpers
// ---------------------------------------------------------------------------
__global__ void fill_f(float* __restrict__ p, float v, int n) {
  int i = blockIdx.x * blockDim.x + threadIdx.x;
  if (i < n) p[i] = v;
}

__global__ void deg_count(const long long* __restrict__ dst, float* __restrict__ deg, int E) {
  int e = blockIdx.x * blockDim.x + threadIdx.x;
  if (e < E) atomicAdd(&deg[(int)dst[e]], 1.0f);
}

__global__ void rsqrt_inplace(float* __restrict__ p, int n) {
  int i = blockIdx.x * blockDim.x + threadIdx.x;
  if (i < n) p[i] = rsqrtf(p[i]);
}

__global__ void cvt_bf16(const float* __restrict__ in, bf16_t* __restrict__ out, long long n) {
  long long i = (long long)blockIdx.x * blockDim.x + threadIdx.x;
  if (i < n) out[i] = f2bf(in[i]);
}

// W is [K, Nout] row-major; produce Wt [Nout, K] row-major in bf16
__global__ void transpose_cvt(const float* __restrict__ W, bf16_t* __restrict__ Wt,
                              int K, int Nout) {
  long long i = (long long)blockIdx.x * blockDim.x + threadIdx.x;
  if (i >= (long long)K * Nout) return;
  int k = (int)(i / Nout);
  int n = (int)(i - (long long)k * Nout);
  Wt[(size_t)n * K + k] = f2bf(W[i]);
}

// out[i,f] = h[i,f] * dinv[i]^2 + bias[f]   (self-loop contribution + bias)
__global__ void init_out(const float* __restrict__ h, const float* __restrict__ dinv,
                         const float* __restrict__ bias, float* __restrict__ out,
                         int Nn, int F) {
  long long idx = (long long)blockIdx.x * blockDim.x + threadIdx.x;
  if (idx >= (long long)Nn * F) return;
  int i = (int)(idx / F);
  int f = (int)(idx - (long long)i * F);
  float di = dinv[i];
  out[idx] = h[idx] * di * di + bias[f];
}

__global__ void relu_cvt(const float* __restrict__ in, bf16_t* __restrict__ out, long long n) {
  long long i = (long long)blockIdx.x * blockDim.x + threadIdx.x;
  if (i < n) {
    float v = in[i];
    out[i] = f2bf(v > 0.0f ? v : 0.0f);
  }
}

// ---------------------------------------------------------------------------
// Edge aggregation: one wave32 per edge, FPL contiguous floats per lane.
// Gather h[src] (coalesced b128/b64 loads) and global_atomic_add_f32 scatter.
// This is the bandwidth-dominant kernel (~1.2 GB total -> ~60 us at 23.3 TB/s).
// ---------------------------------------------------------------------------
template <int FPL>
__global__ void edge_agg(const float* __restrict__ h, const long long* __restrict__ src,
                         const long long* __restrict__ dst, const float* __restrict__ dinv,
                         float* __restrict__ out, int E, int F) {
  long long gid = (long long)blockIdx.x * blockDim.x + threadIdx.x;
  int e = (int)(gid >> 5);
  int lane = (int)(gid & 31);
  if (e >= E) return;
  int s = (int)src[e];
  int d = (int)dst[e];
  float nrm = dinv[s] * dinv[d];
  const float* hp = h + (size_t)s * F + lane * FPL;
  float* op = out + (size_t)d * F + lane * FPL;
  float v[FPL];
#pragma unroll
  for (int j = 0; j < FPL; ++j) v[j] = hp[j];
#pragma unroll
  for (int j = 0; j < FPL; ++j) atomicAdd(op + j, v[j] * nrm);
}

// Fallback for feature dims not a multiple of 32*FPL
__global__ void edge_agg_gen(const float* __restrict__ h, const long long* __restrict__ src,
                             const long long* __restrict__ dst, const float* __restrict__ dinv,
                             float* __restrict__ out, int E, int F) {
  long long gid = (long long)blockIdx.x * blockDim.x + threadIdx.x;
  int e = (int)(gid >> 5);
  int lane = (int)(gid & 31);
  if (e >= E) return;
  int s = (int)src[e];
  int d = (int)dst[e];
  float nrm = dinv[s] * dinv[d];
  for (int f = lane; f < F; f += 32)
    atomicAdd(out + (size_t)d * F + f, h[(size_t)s * F + f] * nrm);
}

// ---------------------------------------------------------------------------
// Dense GEMM: C[M,Nc] = A[M,K] * Bt[Nc,K]^T  (A,Bt bf16 row-major, C f32)
// One wave32 per 16x16 output tile, K-loop of V_WMMA_F32_16X16X32_BF16.
// A-frag (16x32, 16-bit): lane L holds row L%16, K-chunks 8*(L/16) and 16+8*(L/16).
// B-frag identical shape on Bt rows (=output channels).
// C/D: lane L, vgpr r -> (M = r + 8*(L/16), N = L%16).
// Requires K % 32 == 0 and 16B-aligned rows (K % 8 == 0).
// ---------------------------------------------------------------------------
__global__ void gemm_bf16_wmma(const bf16_t* __restrict__ A, const bf16_t* __restrict__ Bt,
                               float* __restrict__ C, int M, int Nc, int K) {
  const int lane = threadIdx.x & 31;
  const int wave = threadIdx.x >> 5;
  const int ntiles = Nc >> 4;
  const int mtiles = (M + 15) >> 4;
  const int tile = blockIdx.x * (blockDim.x >> 5) + wave;
  if (tile >= mtiles * ntiles) return;  // wave-uniform exit: EXEC stays all-ones for WMMA
  const int mt = tile / ntiles;
  const int nt = tile - mt * ntiles;
  const int half = lane >> 4;
  const int lm = lane & 15;

  int arowi = mt * 16 + lm;
  if (arowi >= M) arowi = M - 1;  // clamp (no predication before WMMA)
  const bf16_t* arow = A + (size_t)arowi * K;
  const bf16_t* brow = Bt + (size_t)(nt * 16 + lm) * K;

  v8f c = {};
  for (int k = 0; k < K; k += 32) {
    v8bf a0 = *(const v8bf*)(arow + k + half * 8);
    v8bf a1 = *(const v8bf*)(arow + k + 16 + half * 8);
    v8bf b0 = *(const v8bf*)(brow + k + half * 8);
    v8bf b1 = *(const v8bf*)(brow + k + 16 + half * 8);
    v16bf a, b;
#pragma unroll
    for (int i = 0; i < 8; ++i) {
      a[i] = a0[i]; a[i + 8] = a1[i];
      b[i] = b0[i]; b[i + 8] = b1[i];
    }
    // (neg_a, A, neg_b, B, c_mod, C, reuse_a, reuse_b)
    c = __builtin_amdgcn_wmma_f32_16x16x32_bf16(false, a, false, b, (short)0, c,
                                                false, false);
  }

  const int mbase = mt * 16 + half * 8;
  float* cp = C + (size_t)mbase * Nc + nt * 16 + lm;
#pragma unroll
  for (int r = 0; r < 8; ++r) {
    if (mbase + r < M) cp[(size_t)r * Nc] = c[r];
  }
}

// ---------------------------------------------------------------------------
// Launch: 2-layer GCN
//   inputs: x[N,IN] f32, edge_index[2,E] i64, W1[IN,HID] f32, b1[HID] f32,
//           W2[HID,OUT] f32, b2[OUT] f32 ; output: [N,OUT] f32
// ---------------------------------------------------------------------------
extern "C" void kernel_launch(void* const* d_in, const int* in_sizes, int n_in,
                              void* d_out, int out_size, void* d_ws, size_t ws_size,
                              hipStream_t stream) {
  const float*     x    = (const float*)d_in[0];
  const long long* ei   = (const long long*)d_in[1];
  const float*     W1   = (const float*)d_in[2];
  const float*     b1   = (const float*)d_in[3];
  const float*     W2   = (const float*)d_in[4];
  const float*     b2   = (const float*)d_in[5];

  const int HID = in_sizes[3];                 // 128
  const int OUT = in_sizes[5];                 // 64
  const int INF = in_sizes[2] / HID;           // 128
  const int N   = in_sizes[0] / INF;           // 50000
  const int E   = in_sizes[1] / 2;             // 800000
  const long long* src  = ei;
  const long long* dstp = ei + E;

  // Workspace carve-out (256B aligned):
  char* wsp = (char*)d_ws;
  auto alloc = [&](size_t bytes) -> char* {
    char* p = wsp;
    wsp += (bytes + 255) & ~(size_t)255;
    return p;
  };
  float*  dinv = (float*)alloc((size_t)N * 4);                 // deg -> deg^{-1/2}
  bf16_t* xb   = (bf16_t*)alloc((size_t)N * INF * 2);          // x in bf16
  bf16_t* w1t  = (bf16_t*)alloc((size_t)HID * INF * 2);        // W1^T bf16 [HID,INF]
  bf16_t* w2t  = (bf16_t*)alloc((size_t)OUT * HID * 2);        // W2^T bf16 [OUT,HID]
  float*  h    = (float*)alloc((size_t)N * HID * 4);           // GEMM out (reused L2)
  float*  agg1 = (float*)alloc((size_t)N * HID * 4);           // layer-1 aggregate
  bf16_t* h1b  = (bf16_t*)alloc((size_t)N * HID * 2);          // relu(agg1) bf16
  float*  out  = (float*)d_out;

  const int T = 256;
  auto cdiv = [](long long a, long long b) { return (int)((a + b - 1) / b); };

  // --- degree / normalization ---
  fill_f<<<cdiv(N, T), T, 0, stream>>>(dinv, 1.0f, N);              // self-loop
  deg_count<<<cdiv(E, T), T, 0, stream>>>(dstp, dinv, E);
  rsqrt_inplace<<<cdiv(N, T), T, 0, stream>>>(dinv, N);

  // --- bf16 conversions ---
  cvt_bf16<<<cdiv((long long)N * INF, T), T, 0, stream>>>(x, xb, (long long)N * INF);
  transpose_cvt<<<cdiv((long long)INF * HID, T), T, 0, stream>>>(W1, w1t, INF, HID);
  transpose_cvt<<<cdiv((long long)HID * OUT, T), T, 0, stream>>>(W2, w2t, HID, OUT);

  // --- layer 1: h = X @ W1 (WMMA bf16 -> f32) ---
  {
    int tiles = ((N + 15) / 16) * (HID / 16);
    gemm_bf16_wmma<<<cdiv(tiles, 8), 256, 0, stream>>>(xb, w1t, h, N, HID, INF);
  }
  init_out<<<cdiv((long long)N * HID, T), T, 0, stream>>>(h, dinv, b1, agg1, N, HID);
  {
    long long thr = (long long)E * 32;
    if (HID == 128)
      edge_agg<4><<<cdiv(thr, T), T, 0, stream>>>(h, src, dstp, dinv, agg1, E, HID);
    else if (HID == 64)
      edge_agg<2><<<cdiv(thr, T), T, 0, stream>>>(h, src, dstp, dinv, agg1, E, HID);
    else
      edge_agg_gen<<<cdiv(thr, T), T, 0, stream>>>(h, src, dstp, dinv, agg1, E, HID);
  }
  relu_cvt<<<cdiv((long long)N * HID, T), T, 0, stream>>>(agg1, h1b, (long long)N * HID);

  // --- layer 2: h2 = relu(agg1) @ W2 (reuse h buffer, N*OUT <= N*HID) ---
  {
    int tiles = ((N + 15) / 16) * (OUT / 16);
    gemm_bf16_wmma<<<cdiv(tiles, 8), 256, 0, stream>>>(h1b, w2t, h, N, OUT, HID);
  }
  init_out<<<cdiv((long long)N * OUT, T), T, 0, stream>>>(h, dinv, b2, out, N, OUT);
  {
    long long thr = (long long)E * 32;
    if (OUT == 128)
      edge_agg<4><<<cdiv(thr, T), T, 0, stream>>>(h, src, dstp, dinv, out, E, OUT);
    else if (OUT == 64)
      edge_agg<2><<<cdiv(thr, T), T, 0, stream>>>(h, src, dstp, dinv, out, E, OUT);
    else
      edge_agg_gen<<<cdiv(thr, T), T, 0, stream>>>(h, src, dstp, dinv, out, E, OUT);
  }
}
